// MoEGate_60524679135661
// MI455X (gfx1250) — matrix operand order repaired and checked by reading, not compiled, not run on hardware
//
#include <hip/hip_runtime.h>
#include <hip/hip_bf16.h>

typedef __attribute__((ext_vector_type(16))) _Float16 v16h;
typedef __attribute__((ext_vector_type(8)))  float    v8f;
typedef __attribute__((ext_vector_type(8)))  float    f8v;

#define MG_H   2048
#define MG_E   64
#define MG_S   4096
#define MG_B   4
#define MG_T   16384   // B*S
#define MG_K   6
#define MG_NG  8
#define MG_TKG 3
#define MG_ALPHA 0.001f

// ---------------------------------------------------------------------------
// Kernel 0: convert router weight (E x H) f32 -> f16 into workspace.
// ---------------------------------------------------------------------------
__global__ void mg_w_to_f16(const float* __restrict__ w, _Float16* __restrict__ wf, int n) {
    int i = blockIdx.x * 256 + threadIdx.x;
    if (i < n) wf[i] = (_Float16)w[i];
}

// ---------------------------------------------------------------------------
// Kernel 1: WMMA gemm (16384x2048 @ 2048x64) + softmax + group-limited top-k
// One wave (32 lanes) per block handles 32 tokens.
// ---------------------------------------------------------------------------
__global__ __launch_bounds__(32) void mg_gate_main(
    const float*    __restrict__ x,       // (T,H) f32
    const _Float16* __restrict__ wf,      // (E,H) f16
    float*          __restrict__ out_idx, // (T,K) as float
    float*          __restrict__ out_w,   // (T,K)
    float*          __restrict__ sumsc,   // (B,E) score-sum accumulators
    float*          __restrict__ cnt)     // (B,E) pick-count accumulators
{
    __shared__ float lds[32 * 65 + 64];   // 32 tokens x 64 experts (stride 65) + hist
    float* hist = lds + 32 * 65;

    const int lane    = threadIdx.x;
    const int lane_lo = lane & 15;
    const int lane_hi = lane >> 4;
    const int t0      = blockIdx.x * 32;
    const int b       = t0 >> 12;         // t0 / S, 32 | 4096 so no straddle

    v8f acc[2][4];
    #pragma unroll
    for (int mt = 0; mt < 2; ++mt)
        #pragma unroll
        for (int nt = 0; nt < 4; ++nt)
            acc[mt][nt] = (v8f){0.f,0.f,0.f,0.f,0.f,0.f,0.f,0.f};

    // A-fragment addressing (contiguous k order used here, not the ISA table order):
    //   lane holds row M = lane_lo, K = lane_hi*16 + 0..15 contiguous
    const float* xrow0 = x + (size_t)(t0 + lane_lo) * MG_H;
    const float* xrow1 = xrow0 + (size_t)16 * MG_H;
    const int abase = lane_hi * 16;
    // B-fragment addressing (same contiguous k order as A, not the ISA table order):
    //   lane holds col N = lane_lo (expert), K = lane_hi*16 + 0..15 contiguous
    const int kb = lane_hi * 16;

    #pragma unroll 2
    for (int k0 = 0; k0 < MG_H; k0 += 32) {
        v16h af0, af1;
        {
            const float* p0 = xrow0 + k0 + abase;
            f8v c1 = *(const f8v*)(p0);
            f8v c2 = *(const f8v*)(p0 + 8);
            #pragma unroll
            for (int i = 0; i < 8; ++i) { af0[i] = (_Float16)c1[i]; af0[i+8] = (_Float16)c2[i]; }
            const float* p1 = xrow1 + k0 + abase;
            f8v d1 = *(const f8v*)(p1);
            f8v d2 = *(const f8v*)(p1 + 8);
            #pragma unroll
            for (int i = 0; i < 8; ++i) { af1[i] = (_Float16)d1[i]; af1[i+8] = (_Float16)d2[i]; }
        }
        #pragma unroll
        for (int nt = 0; nt < 4; ++nt) {
            const _Float16* wp = wf + (size_t)(nt * 16 + lane_lo) * MG_H + k0 + kb;
            v16h bf = *(const v16h*)wp;
            acc[0][nt] = __builtin_amdgcn_wmma_f32_16x16x32_f16(
                false, af0, false, bf, (short)0, acc[0][nt], false, false);
            acc[1][nt] = __builtin_amdgcn_wmma_f32_16x16x32_f16(
                false, af1, false, bf, (short)0, acc[1][nt], false, false);
        }
    }

    // Spill logits to LDS, transposing to token-major (stride 65 => conflict-free reads).
    // C layout: VGPR j, lanes 0-15: M=j, lanes 16-31: M=j+8; N = lane_lo.
    #pragma unroll
    for (int mt = 0; mt < 2; ++mt)
        #pragma unroll
        for (int nt = 0; nt < 4; ++nt)
            #pragma unroll
            for (int j = 0; j < 8; ++j) {
                int tok = mt * 16 + 8 * lane_hi + j;
                int e   = nt * 16 + lane_lo;
                lds[tok * 65 + e] = acc[mt][nt][j];
            }
    __syncthreads();

    hist[lane] = 0.f;
    hist[lane + 32] = 0.f;
    __syncthreads();

    // ---- gating: one lane per token ----
    float sc[64];
    #pragma unroll
    for (int e = 0; e < 64; ++e) sc[e] = lds[lane * 65 + e];

    float mx = sc[0];
    #pragma unroll
    for (int e = 1; e < 64; ++e) mx = fmaxf(mx, sc[e]);
    float ssum = 0.f;
    #pragma unroll
    for (int e = 0; e < 64; ++e) { sc[e] = __expf(sc[e] - mx); ssum += sc[e]; }
    float inv = 1.0f / ssum;
    #pragma unroll
    for (int e = 0; e < 64; ++e) sc[e] *= inv;

    // scores back to LDS for the aux-loss reduction (before destructive top-k)
    #pragma unroll
    for (int e = 0; e < 64; ++e) lds[lane * 65 + e] = sc[e];

    // group maxes over 8 groups of 8
    float gm[8];
    #pragma unroll
    for (int g = 0; g < 8; ++g) {
        float m = sc[g * 8];
        #pragma unroll
        for (int j = 1; j < 8; ++j) m = fmaxf(m, sc[g * 8 + j]);
        gm[g] = m;
    }
    // top-3 groups (lowest index wins ties, matching jax top_k)
    unsigned gbits = 0;
    #pragma unroll
    for (int p = 0; p < MG_TKG; ++p) {
        float bg = -1.f; int bgi = 0;
        #pragma unroll
        for (int g = 0; g < 8; ++g) {
            bool freeg = !((gbits >> g) & 1u);
            if (freeg && gm[g] > bg) { bg = gm[g]; bgi = g; }
        }
        gbits |= 1u << bgi;
    }
    // mask disallowed experts
    float msc[64];
    #pragma unroll
    for (int e = 0; e < 64; ++e)
        msc[e] = ((gbits >> (e >> 3)) & 1u) ? sc[e] : -1.f;

    // top-6 of masked scores via picked bitmask
    unsigned long long picked = 0ull;
    int   ti[MG_K];
    float tw[MG_K];
    float wsum = 1e-20f;
    #pragma unroll
    for (int i = 0; i < MG_K; ++i) {
        float best = -2.f; int bi = 0;
        #pragma unroll
        for (int e = 0; e < 64; ++e) {
            float v = ((picked >> e) & 1ull) ? -2.f : msc[e];
            if (v > best) { best = v; bi = e; }
        }
        ti[i] = bi; tw[i] = best; wsum += best;
        picked |= (1ull << bi);
    }

    float invw = 1.0f / wsum;
    size_t obase = (size_t)(t0 + lane) * MG_K;
    #pragma unroll
    for (int i = 0; i < MG_K; ++i) {
        out_idx[obase + i] = (float)ti[i];
        out_w[obase + i]   = tw[i] * invw;
    }
    #pragma unroll
    for (int i = 0; i < MG_K; ++i) atomicAdd(&hist[ti[i]], 1.0f);

    __syncthreads();

    // per-wave reduction of scores over 32 tokens; 2 experts per lane
    int e0 = lane, e1 = lane + 32;
    float s0 = 0.f, s1 = 0.f;
    #pragma unroll
    for (int tok = 0; tok < 32; ++tok) {
        s0 += lds[tok * 65 + e0];
        s1 += lds[tok * 65 + e1];
    }
    atomicAdd(&sumsc[b * MG_E + e0], s0);
    atomicAdd(&sumsc[b * MG_E + e1], s1);
    atomicAdd(&cnt[b * MG_E + e0], hist[e0]);
    atomicAdd(&cnt[b * MG_E + e1], hist[e1]);
}

// ---------------------------------------------------------------------------
// Kernel 2: aux_loss = mean_b( sum_e ce[b,e]*mean_scores[b,e] ) * ALPHA
// ---------------------------------------------------------------------------
__global__ void mg_finalize(const float* __restrict__ sumsc,
                            const float* __restrict__ cnt,
                            float* __restrict__ aux_out)
{
    __shared__ float red[256];
    int tid = threadIdx.x;           // tid = b*64 + e
    float ce = cnt[tid] * ((float)MG_E / ((float)MG_S * (float)MG_K));
    float ms = sumsc[tid] * (1.0f / (float)MG_S);
    red[tid] = ce * ms;
    __syncthreads();
    for (int s = 128; s > 0; s >>= 1) {
        if (tid < s) red[tid] += red[tid + s];
        __syncthreads();
    }
    if (tid == 0) aux_out[0] = red[0] * (MG_ALPHA / (float)MG_B);
}

// ---------------------------------------------------------------------------
extern "C" void kernel_launch(void* const* d_in, const int* in_sizes, int n_in,
                              void* d_out, int out_size, void* d_ws, size_t ws_size,
                              hipStream_t stream) {
    const float* x = (const float*)d_in[0];   // (B,S,H) f32
    const float* w = (const float*)d_in[1];   // (E,H)   f32
    float* out = (float*)d_out;               // [T*K idx | T*K weight | aux]

    _Float16* wf    = (_Float16*)d_ws;                                  // 256 KB
    float*    sumsc = (float*)((char*)d_ws + (size_t)MG_E * MG_H * 2);  // 1 KB
    float*    cntp  = sumsc + MG_B * MG_E;                              // 1 KB

    hipMemsetAsync(sumsc, 0, 2 * MG_B * MG_E * sizeof(float), stream);

    mg_w_to_f16<<<(MG_E * MG_H + 255) / 256, 256, 0, stream>>>(w, wf, MG_E * MG_H);

    mg_gate_main<<<MG_T / 32, 32, 0, stream>>>(
        x, wf,
        out,                         // topk_idx as float
        out + (size_t)MG_T * MG_K,   // topk_weight
        sumsc, cntp);

    mg_finalize<<<1, 256, 0, stream>>>(sumsc, cntp, out + (size_t)2 * MG_T * MG_K);
}
